// GCNLayer_44315472560382
// MI455X (gfx1250) — matrix operand order repaired and checked by reading, not compiled, not run on hardware
//
#include <hip/hip_runtime.h>
#include <hip/hip_bf16.h>

// ---------- types ----------
typedef __attribute__((ext_vector_type(16))) _Float16 v16h;
typedef __attribute__((ext_vector_type(8)))  float    v8f;
typedef __attribute__((ext_vector_type(4)))  unsigned int u32x4;
typedef __attribute__((ext_vector_type(4)))  float    f32x4;
typedef __attribute__((ext_vector_type(4)))  int      v4i_;

union Frag16 { v16h h; u32x4 q[2]; };

#define B_ 8
#define L_ 2048
#define D_ 512
#define TM 128
#define TN 128
#define TK 64
#define LDT 72   // padded LDS row stride (f16 elems); 144B, 16B-aligned

// ---------- gfx1250 async global->LDS (guarded; fallback = reg-staged) ----
#if defined(__HIP_DEVICE_COMPILE__) && __has_builtin(__builtin_amdgcn_global_load_async_to_lds_b128)
#define HAVE_ASYNC 1
#else
#define HAVE_ASYNC 0
#endif

typedef __attribute__((address_space(1))) v4i_ gv4i;   // global (device) int4
typedef __attribute__((address_space(3))) v4i_ lv4i;   // LDS int4

__device__ __forceinline__ void async_cp16(const void* g, void* l) {
#if HAVE_ASYNC
    __builtin_amdgcn_global_load_async_to_lds_b128((gv4i*)g, (lv4i*)l, 0, 0);
#else
    (void)g; (void)l;
#endif
}
__device__ __forceinline__ void async_wait() {
#if HAVE_ASYNC
#if __has_builtin(__builtin_amdgcn_s_wait_asynccnt)
    __builtin_amdgcn_s_wait_asynccnt(0);
#else
    asm volatile("s_wait_asynccnt 0" ::: "memory");
#endif
#endif
}

// ---------- shared WMMA tile compute: 8 waves, each 32x64 of a 128x128 tile
__device__ __forceinline__ void wmma_tile(const _Float16 (&Asb)[TM][LDT],
                                          const _Float16 (&Bsb)[TN][LDT],
                                          int lane, int wm, int wn, v8f acc[2][4])
{
    const int ka = (lane & 16) ? 8 : 0;    // A frag: K 0-7 | 8-15 (+16 hi half)
    const int kb = (lane & 16) ? 8 : 0;    // B frag: K 0-7 | 8-15 (+16 hi half)
    const int rr = lane & 15;
    for (int kk = 0; kk < TK; kk += 32) {
        Frag16 af[2], bf[4];
        for (int mi = 0; mi < 2; ++mi) {
            int m = wm + mi * 16 + rr;
            af[mi].q[0] = *reinterpret_cast<const u32x4*>(&Asb[m][kk + ka]);
            af[mi].q[1] = *reinterpret_cast<const u32x4*>(&Asb[m][kk + ka + 16]);
        }
        for (int ni = 0; ni < 4; ++ni) {
            int n = wn + ni * 16 + rr;
            bf[ni].q[0] = *reinterpret_cast<const u32x4*>(&Bsb[n][kk + kb]);
            bf[ni].q[1] = *reinterpret_cast<const u32x4*>(&Bsb[n][kk + kb + 16]);
        }
        for (int mi = 0; mi < 2; ++mi)
            for (int ni = 0; ni < 4; ++ni)
                acc[mi][ni] = __builtin_amdgcn_wmma_f32_16x16x32_f16(
                    false, af[mi].h, false, bf[ni].h, (short)0, acc[mi][ni], false, false);
    }
}

// =====================================================================
// Kernel 1: degrees -> p = valid * deg^-1/2, q = 1/deg
// =====================================================================
__global__ __launch_bounds__(256)
void deg_kernel(const float* __restrict__ adj, const unsigned char* __restrict__ pad,
                float* __restrict__ p, float* __restrict__ q)
{
    __shared__ float red[256];
    const int row = blockIdx.x;            // b*L + i
    const int b   = row >> 11;
    const int tid = threadIdx.x;
    const float* arow = adj + (size_t)row * L_;
    const unsigned char* prow = pad + ((size_t)b << 11);

    float s = 0.f;
    for (int j = tid; j < L_; j += 256)
        s += arow[j] * (prow[j] ? 0.f : 1.f);
    red[tid] = s;
    __syncthreads();
    for (int off = 128; off > 0; off >>= 1) {
        if (tid < off) red[tid] += red[tid + off];
        __syncthreads();
    }
    if (tid == 0) {
        float vi  = pad[row] ? 0.f : 1.f;
        float deg = vi * red[0] + 1.0f;
        float dis = rsqrtf(deg);
        p[row] = vi * dis;
        q[row] = 1.0f / deg;
    }
}

// =====================================================================
// Kernel 2: xs[b,d,l] = f16( p[b,l] * x[b,l,d] )   (col-scale folded in)
// =====================================================================
__global__ __launch_bounds__(256)
void cvt_x_kernel(const float* __restrict__ x, const float* __restrict__ p,
                  _Float16* __restrict__ xs)
{
    __shared__ float t[32][33];
    const int b  = blockIdx.z;
    const int l0 = blockIdx.y * 32;
    const int d0 = blockIdx.x * 32;
    const int tx = threadIdx.x & 31;
    const int ty = threadIdx.x >> 5;     // 0..7
    for (int i = 0; i < 4; ++i) {
        int l = l0 + ty + i * 8;
        t[ty + i * 8][tx] = x[((size_t)b * L_ + l) * D_ + d0 + tx];
    }
    __syncthreads();
    const float pl = p[b * L_ + l0 + tx];
    for (int i = 0; i < 4; ++i) {
        int d = d0 + ty + i * 8;
        xs[((size_t)b * D_ + d) * L_ + l0 + tx] = (_Float16)(pl * t[tx][ty + i * 8]);
    }
}

// =====================================================================
// Kernel 3: W f32 -> f16 (row = out, col = in : already K-contiguous)
// =====================================================================
__global__ __launch_bounds__(256)
void cvt_w_kernel(const float* __restrict__ W, _Float16* __restrict__ Wb)
{
    int g = blockIdx.x * 256 + threadIdx.x;
    Wb[g] = (_Float16)W[g];
}

// =====================================================================
// Kernel 4: acc = Adj @ xs ; agg = ew*(p_i*acc + q_i*x_i)  -> f16 [B,L,D]
// Double-buffered LDS; B-tile via async DMA when available.
// =====================================================================
__global__ __launch_bounds__(256)
void gemm1_kernel(const float* __restrict__ adj, const _Float16* __restrict__ xs,
                  const float* __restrict__ p, const float* __restrict__ q,
                  const float* __restrict__ ewp, const float* __restrict__ x,
                  _Float16* __restrict__ agg)
{
    __shared__ _Float16 As[2][TM][LDT];
    __shared__ _Float16 Bs[2][TN][LDT];
    __shared__ float pis[TM], qis[TM];

    const int b   = blockIdx.z;
    const int i0  = blockIdx.y * TM;
    const int n0  = blockIdx.x * TN;
    const int tid = threadIdx.x;
    const int lane = tid & 31, wid = tid >> 5;
    const int wm = (wid & 3) * 32;
    const int wn = (wid >> 2) * 64;
    const float ew = ewp[0];

    if (tid < TM) { pis[tid] = p[b * L_ + i0 + tid]; qis[tid] = q[b * L_ + i0 + tid]; }

    v8f acc[2][4];
    for (int mi = 0; mi < 2; ++mi)
        for (int ni = 0; ni < 4; ++ni)
            for (int r = 0; r < 8; ++r) acc[mi][ni][r] = 0.f;

    const size_t adj_base = ((size_t)b * L_ + i0) * L_;
    const size_t xs_base  = ((size_t)b * D_ + n0) * (size_t)L_;

    f32x4 aregs[8];
#if !HAVE_ASYNC
    u32x4 bregs[4];
#endif

    // ---- loaders (fixed per-thread coords; m/n depend on iteration) ----
    auto loadA = [&](int k0) {
        for (int it = 0; it < 8; ++it) {
            int idx = it * 256 + tid, m = idx >> 4, k4 = (idx & 15) * 4;
            aregs[it] = *reinterpret_cast<const f32x4*>(&adj[adj_base + (size_t)m * L_ + k0 + k4]);
        }
    };
    auto storeA = [&](int buf) {
        for (int it = 0; it < 8; ++it) {
            int idx = it * 256 + tid, m = idx >> 4, k4 = (idx & 15) * 4;
            As[buf][m][k4 + 0] = (_Float16)aregs[it].x;
            As[buf][m][k4 + 1] = (_Float16)aregs[it].y;
            As[buf][m][k4 + 2] = (_Float16)aregs[it].z;
            As[buf][m][k4 + 3] = (_Float16)aregs[it].w;
        }
    };
    auto loadB = [&](int k0, int buf) {
        for (int it = 0; it < 4; ++it) {
            int idx = it * 256 + tid, n = idx >> 3, k8 = (idx & 7) * 8;
#if HAVE_ASYNC
            async_cp16(&xs[xs_base + (size_t)n * L_ + k0 + k8], &Bs[buf][n][k8]);
#else
            bregs[it] = *reinterpret_cast<const u32x4*>(&xs[xs_base + (size_t)n * L_ + k0 + k8]);
#endif
        }
    };
#if !HAVE_ASYNC
    auto storeB = [&](int buf) {
        for (int it = 0; it < 4; ++it) {
            int idx = it * 256 + tid, n = idx >> 3, k8 = (idx & 7) * 8;
            *reinterpret_cast<u32x4*>(&Bs[buf][n][k8]) = bregs[it];
        }
    };
#endif

    // ---- prologue: tile 0 ----
    loadB(0, 0);
    loadA(0);
    storeA(0);
#if !HAVE_ASYNC
    storeB(0);
#endif
    async_wait();
    __syncthreads();

    const int NT = L_ / TK;   // 32
    for (int t = 0; t < NT; ++t) {
        const int cur = t & 1, nxt = cur ^ 1;
        const bool more = (t + 1 < NT);
        if (more) { loadB((t + 1) * TK, nxt); loadA((t + 1) * TK); }
        wmma_tile(As[cur], Bs[cur], lane, wm, wn, acc);   // hides next-tile latency
        if (more) {
            storeA(nxt);
#if !HAVE_ASYNC
            storeB(nxt);
#endif
        }
        async_wait();
        __syncthreads();
    }

    // ---- epilogue: row scale + diagonal term in f32 ----
    const int rr  = lane & 15;
    const int mhi = (lane & 16) ? 8 : 0;
    for (int mi = 0; mi < 2; ++mi)
        for (int ni = 0; ni < 4; ++ni)
            for (int r = 0; r < 8; ++r) {
                int lm = wm + mi * 16 + mhi + r;
                int gm = i0 + lm;
                int gn = n0 + wn + ni * 16 + rr;
                float xv = x[((size_t)b * L_ + gm) * D_ + gn];
                float v  = ew * (pis[lm] * acc[mi][ni][r] + qis[lm] * xv);
                agg[((size_t)b * L_ + gm) * D_ + gn] = (_Float16)v;
            }
}

// =====================================================================
// Kernel 5: out2 = relu(agg @ W^T + b) * D^-1/2  -> f16 [B*L, D]
// Both tiles are pure f16 copies -> async DMA + double buffer.
// =====================================================================
__global__ __launch_bounds__(256)
void gemm2_kernel(const _Float16* __restrict__ agg, const _Float16* __restrict__ Wb,
                  const float* __restrict__ bias, _Float16* __restrict__ out2)
{
    __shared__ _Float16 As[2][TM][LDT];
    __shared__ _Float16 Bs[2][TN][LDT];

    const int m0  = blockIdx.y * TM;      // rows over B*L
    const int n0  = blockIdx.x * TN;      // cols (D out)
    const int tid = threadIdx.x;
    const int lane = tid & 31, wid = tid >> 5;
    const int wm = (wid & 3) * 32;
    const int wn = (wid >> 2) * 64;

    v8f acc[2][4];
    for (int mi = 0; mi < 2; ++mi)
        for (int ni = 0; ni < 4; ++ni)
            for (int r = 0; r < 8; ++r) acc[mi][ni][r] = 0.f;

#if !HAVE_ASYNC
    u32x4 ar[4], br[4];
#endif
    auto loadAB = [&](int k0, int buf) {
        for (int it = 0; it < 4; ++it) {
            int idx = it * 256 + tid, m = idx >> 3, k8 = (idx & 7) * 8;
#if HAVE_ASYNC
            async_cp16(&agg[(size_t)(m0 + m) * D_ + k0 + k8], &As[buf][m][k8]);
            async_cp16(&Wb [(size_t)(n0 + m) * D_ + k0 + k8], &Bs[buf][m][k8]);
#else
            ar[it] = *reinterpret_cast<const u32x4*>(&agg[(size_t)(m0 + m) * D_ + k0 + k8]);
            br[it] = *reinterpret_cast<const u32x4*>(&Wb [(size_t)(n0 + m) * D_ + k0 + k8]);
#endif
        }
    };
#if !HAVE_ASYNC
    auto storeAB = [&](int buf) {
        for (int it = 0; it < 4; ++it) {
            int idx = it * 256 + tid, m = idx >> 3, k8 = (idx & 7) * 8;
            *reinterpret_cast<u32x4*>(&As[buf][m][k8]) = ar[it];
            *reinterpret_cast<u32x4*>(&Bs[buf][m][k8]) = br[it];
        }
    };
#endif

    loadAB(0, 0);
#if !HAVE_ASYNC
    storeAB(0);
#endif
    async_wait();
    __syncthreads();

    const int NT = D_ / TK;   // 8
    for (int t = 0; t < NT; ++t) {
        const int cur = t & 1, nxt = cur ^ 1;
        const bool more = (t + 1 < NT);
        if (more) loadAB((t + 1) * TK, nxt);
        wmma_tile(As[cur], Bs[cur], lane, wm, wn, acc);
#if !HAVE_ASYNC
        if (more) storeAB(nxt);
#endif
        async_wait();
        __syncthreads();
    }

    const float scale = 0.044194173824159216f;   // 512^-0.5
    const int rr  = lane & 15;
    const int mhi = (lane & 16) ? 8 : 0;
    for (int mi = 0; mi < 2; ++mi)
        for (int ni = 0; ni < 4; ++ni)
            for (int r = 0; r < 8; ++r) {
                int gm = m0 + wm + mi * 16 + mhi + r;
                int gn = n0 + wn + ni * 16 + rr;
                float v = acc[mi][ni][r] + bias[gn];
                v = fmaxf(v, 0.f) * scale;
                out2[(size_t)gm * D_ + gn] = (_Float16)v;
            }
}

// =====================================================================
// Kernel 6: h = x + out2; layernorm over D -> f32 d_out
// =====================================================================
__global__ __launch_bounds__(256)
void ln_kernel(const float* __restrict__ x, const _Float16* __restrict__ out2,
               const float* __restrict__ lnw, const float* __restrict__ lnb,
               float* __restrict__ out)
{
    __shared__ float s1[256], s2[256];
    __shared__ float mu_s, rstd_s;
    const size_t base = (size_t)blockIdx.x * D_;
    const int tid = threadIdx.x;
    const int c0 = tid, c1 = tid + 256;

    float h0 = x[base + c0] + (float)out2[base + c0];
    float h1 = x[base + c1] + (float)out2[base + c1];
    s1[tid] = h0 + h1;
    s2[tid] = h0 * h0 + h1 * h1;
    __syncthreads();
    for (int off = 128; off > 0; off >>= 1) {
        if (tid < off) { s1[tid] += s1[tid + off]; s2[tid] += s2[tid + off]; }
        __syncthreads();
    }
    if (tid == 0) {
        float mu  = s1[0] * (1.0f / D_);
        float var = s2[0] * (1.0f / D_) - mu * mu;
        mu_s = mu;
        rstd_s = rsqrtf(var + 1e-5f);
    }
    __syncthreads();
    float mu = mu_s, rstd = rstd_s;
    out[base + c0] = lnw[c0] * (h0 - mu) * rstd + lnb[c0];
    out[base + c1] = lnw[c1] * (h1 - mu) * rstd + lnb[c1];
}

// =====================================================================
extern "C" void kernel_launch(void* const* d_in, const int* in_sizes, int n_in,
                              void* d_out, int out_size, void* d_ws, size_t ws_size,
                              hipStream_t stream)
{
    const float*         x    = (const float*)d_in[0];
    const float*         adj  = (const float*)d_in[1];
    const unsigned char* pad  = (const unsigned char*)d_in[2];
    const float*         W    = (const float*)d_in[3];
    const float*         bias = (const float*)d_in[4];
    const float*         lnw  = (const float*)d_in[5];
    const float*         lnb  = (const float*)d_in[6];
    const float*         ew   = (const float*)d_in[7];
    float*               out  = (float*)d_out;

    // workspace layout (16B-aligned): p,q f32; xs,Wb,agg,out2 f16
    char* ws = (char*)d_ws;
    float*    p    = (float*)ws;                                   //  64 KB
    float*    q    = (float*)(ws + (size_t)64 * 1024);             //  64 KB
    _Float16* xs   = (_Float16*)(ws + (size_t)128 * 1024);         //  16 MB
    _Float16* Wb   = (_Float16*)(ws + (size_t)128 * 1024 + (size_t)16 * 1024 * 1024);
    _Float16* agg  = (_Float16*)(ws + (size_t)128 * 1024 + (size_t)16 * 1024 * 1024 + (size_t)512 * 1024);
    _Float16* out2 = (_Float16*)(ws + (size_t)128 * 1024 + (size_t)32 * 1024 * 1024 + (size_t)512 * 1024);

    deg_kernel  <<<B_ * L_, 256, 0, stream>>>(adj, pad, p, q);
    cvt_x_kernel<<<dim3(D_ / 32, L_ / 32, B_), 256, 0, stream>>>(x, p, xs);
    cvt_w_kernel<<<(D_ * D_) / 256, 256, 0, stream>>>(W, Wb);
    gemm1_kernel<<<dim3(D_ / TN, L_ / TM, B_), 256, 0, stream>>>(adj, xs, p, q, ew, x, agg);
    gemm2_kernel<<<dim3(D_ / TN, (B_ * L_) / TM), 256, 0, stream>>>(agg, Wb, bias, out2);
    ln_kernel   <<<B_ * L_, 256, 0, stream>>>(x, out2, lnw, lnb, out);
}
